// TransformerEncoder_18391049961502
// MI455X (gfx1250) — matrix operand order repaired and checked by reading, not compiled, not run on hardware
//
#include <hip/hip_runtime.h>

typedef __attribute__((ext_vector_type(16))) _Float16 v16h;
typedef __attribute__((ext_vector_type(8)))  float    v8f;
typedef __attribute__((ext_vector_type(4)))  unsigned int u32x4;
typedef __attribute__((ext_vector_type(8)))  int i32x8;
typedef __attribute__((ext_vector_type(4)))  int i32x4;

union H2 { unsigned int u; _Float16 h[2]; };

__device__ inline v8f v8f_zero() { v8f z = {0.f,0.f,0.f,0.f,0.f,0.f,0.f,0.f}; return z; }

__device__ inline v8f wmma16(v16h a, v16h b, v8f c) {
  return __builtin_amdgcn_wmma_f32_16x16x32_f16(false, a, false, b, (short)0, c, false, false);
}

// ---------------------------------------------------------------- TDM tile load
// DMA a 2D tile (rows x 64 f16 elements, global row stride = stride_elems) into
// LDS with 16B padding inserted after every 128B row -> padded LDS stride of
// 72 halves (= LDT below). Issued per-wave (EXEC ignored); caller guards to one
// wave and waits on TENSORcnt.
__device__ inline void tdm_load_tile(const _Float16* gsrc, _Float16* lds_dst,
                                     int rows, int stride_elems) {
  unsigned lds_addr = (unsigned)__builtin_amdgcn_readfirstlane((int)(size_t)lds_dst);
  unsigned long long ga = (unsigned long long)(size_t)gsrc;
  unsigned ga_lo = (unsigned)__builtin_amdgcn_readfirstlane((int)(unsigned)(ga & 0xFFFFFFFFull));
  unsigned ga_hi = (unsigned)__builtin_amdgcn_readfirstlane((int)(unsigned)(ga >> 32));

  u32x4 g0;
  g0[0] = 1u;                                   // count = 1 valid descriptor
  g0[1] = lds_addr;                             // lds_addr [63:32]
  g0[2] = ga_lo;                                // global_addr [95:64]
  g0[3] = (ga_hi & 0x01FFFFFFu) | (2u << 30);   // global_addr hi bits, type=2 (image)

  i32x8 g1;
  g1[0] = (1 << 16)      // data_size = 2 bytes
        | (1 << 20)      // pad_enable
        | (4 << 22)      // pad_interval: 2^(4+1)=32 DWORDs (=128B row)
        | (3 << 25);     // pad_amount: 4 DWORDs (=16B -> LDT = 72 halves)
  g1[1] = (64 << 16);                 // tensor_dim0 = 64 (bits 79:48 lo)
  g1[2] = ((rows & 0xFFFF) << 16);    // tensor_dim0 hi=0 | tensor_dim1 = rows
  g1[3] = (64 << 16);                 // tensor_dim1 hi=0 | tile_dim0 = 64
  g1[4] = (rows & 0xFFFF);            // tile_dim1 = rows | tile_dim2 = 0 (2D)
  g1[5] = stride_elems;               // tensor_dim0_stride lo32 (data_size units)
  g1[6] = 0;                          // stride hi | tensor_dim1_stride lo (unused, 2D)
  g1[7] = 0;

  i32x4 z4; z4[0] = 0; z4[1] = 0; z4[2] = 0; z4[3] = 0;
#if __clang_major__ >= 23
  i32x8 z8; z8[0]=0; z8[1]=0; z8[2]=0; z8[3]=0; z8[4]=0; z8[5]=0; z8[6]=0; z8[7]=0;
  __builtin_amdgcn_tensor_load_to_lds(g0, g1, z4, z4, z8, 0);
#else
  __builtin_amdgcn_tensor_load_to_lds(g0, g1, z4, z4, 0);
#endif
}

__device__ inline void tdm_wait() {
  __builtin_amdgcn_s_wait_tensorcnt((short)0);
}

// A-fragment (16x32 f16). p points at row0 of the 16-row tile, already offset to
// the K-chunk base. ldh = row stride in halves. Lane l (0..15): row M=l; half=l>>4
// selects K groups: K(e) = e + half*16.
__device__ inline v16h load_a_frag(const _Float16* __restrict__ p, int ldh) {
  int lane = threadIdx.x & 31;
  int half = lane >> 4;
  int m    = lane & 15;
  const _Float16* q = p + m * ldh;
  v16h a;
#pragma unroll
  for (int pr = 0; pr < 8; ++pr) {
    int kb = 2 * pr + half * 16;   // even -> 4B aligned
    H2 t; t.u = *(const unsigned int*)(q + kb);
    a[2 * pr] = t.h[0]; a[2 * pr + 1] = t.h[1];
  }
  return a;
}

// B-fragment (32x16 f16) read from a TRANSPOSED [N][K] tile: lane holds column
// n = lane&15; K(e) = e + half*16 -> contiguous pairs along K.
__device__ inline v16h load_bT_frag(const _Float16* __restrict__ p, int ldh) {
  int lane = threadIdx.x & 31;
  int half = lane >> 4;
  int n    = lane & 15;
  const _Float16* q = p + n * ldh;
  v16h b;
#pragma unroll
  for (int pr = 0; pr < 8; ++pr) {
    int kb = 2 * pr + half * 16;
    H2 t; t.u = *(const unsigned int*)(q + kb);
    b[2 * pr] = t.h[0]; b[2 * pr + 1] = t.h[1];
  }
  return b;
}

// B-fragment from a [K][N] row-major tile (used for p@v where v is [t][c]).
__device__ inline v16h load_bKN_frag(const _Float16* __restrict__ lds, int k0, int n0, int ldh) {
  int lane = threadIdx.x & 31;
  int half = lane >> 4;
  int n    = n0 + (lane & 15);
  v16h b;
#pragma unroll
  for (int e = 0; e < 16; ++e) {
    int k = k0 + e + half * 16;
    b[e] = lds[k * ldh + n];
  }
  return b;
}

// ---------------------------------------------------------------- weight convert
// dst[b][c][r] = src[b][r][c]  (fp32 -> f16 transposed)
__global__ void convert_T_kernel(const float* __restrict__ src, _Float16* __restrict__ dst,
                                 int batch, int rows, int cols) {
  size_t n = (size_t)batch * rows * cols;
  size_t stride = (size_t)gridDim.x * blockDim.x;
  for (size_t i = (size_t)blockIdx.x * blockDim.x + threadIdx.x; i < n; i += stride) {
    size_t rc  = (size_t)rows * cols;
    size_t bb  = i / rc;
    size_t rem = i - bb * rc;
    int r = (int)(rem / cols), c = (int)(rem % cols);
    dst[bb * rc + (size_t)c * rows + r] = (_Float16)src[i];
  }
}

// ---------------------------------------------------------------- LayerNorm -> f16
__global__ void __launch_bounds__(256) ln_kernel(const float* __restrict__ x,
                                                 const float* __restrict__ g,
                                                 const float* __restrict__ bta,
                                                 _Float16* __restrict__ out, int D) {
  int row = blockIdx.x;
  const float* xr = x + (size_t)row * D;
  int tid = threadIdx.x;
  float s = 0.f, s2 = 0.f;
  for (int c = tid; c < D; c += 256) { float t = xr[c]; s += t; s2 += t * t; }
#pragma unroll
  for (int off = 1; off < 32; off <<= 1) { s += __shfl_xor(s, off, 32); s2 += __shfl_xor(s2, off, 32); }
  __shared__ float swsum[8], swsq[8], sstat[2];
  if ((tid & 31) == 0) { swsum[tid >> 5] = s; swsq[tid >> 5] = s2; }
  __syncthreads();
  if (tid == 0) {
    float a = 0.f, b2 = 0.f;
#pragma unroll
    for (int i = 0; i < 8; ++i) { a += swsum[i]; b2 += swsq[i]; }
    float mean = a / D;
    float var  = b2 / D - mean * mean;
    sstat[0] = mean; sstat[1] = rsqrtf(var + 1e-5f);
  }
  __syncthreads();
  float mean = sstat[0], rstd = sstat[1];
  for (int c = tid; c < D; c += 256)
    out[(size_t)row * D + c] = (_Float16)((xr[c] - mean) * rstd * g[c] + bta[c]);
}

// ---------------------------------------------------------------- QKV projection
// grid: (S/64, B*H), block 128 (4 waves, 16 token-rows each).
__global__ void __launch_bounds__(128) qkv_kernel(
    const _Float16* __restrict__ n1,                        // [B*S][D] f16
    const _Float16* __restrict__ WqT, const _Float16* __restrict__ WkT,
    const _Float16* __restrict__ WvT,                       // [H][64(cout)][64(cin)]
    const float* __restrict__ bq, const float* __restrict__ bk, const float* __restrict__ bv,
    _Float16* __restrict__ q, _Float16* __restrict__ k, _Float16* __restrict__ v, // [B*H][S][64]
    int B, int H, int S, int D) {
  const int LDA = 72;
  __shared__ _Float16 sX[64 * LDA];
  __shared__ _Float16 sW[3 * 64 * LDA];
  int bh = blockIdx.y;
  int b  = bh / H, h = bh % H;
  int s0 = blockIdx.x * 64;
  int tid = threadIdx.x, wave = tid >> 5, lane = tid & 31, half = lane >> 4, ln = lane & 15;

  if (wave == 0) {                               // TDM-stage X slice + 3 weight mats
    tdm_load_tile(n1 + (size_t)(b * S + s0) * D + h * 64, sX, 64, D);
    tdm_load_tile(WqT + (size_t)h * 4096, sW + 0 * 64 * LDA, 64, 64);
    tdm_load_tile(WkT + (size_t)h * 4096, sW + 1 * 64 * LDA, 64, 64);
    tdm_load_tile(WvT + (size_t)h * 4096, sW + 2 * 64 * LDA, 64, 64);
    tdm_wait();
  }
  __syncthreads();

  v16h af[2];
#pragma unroll
  for (int kc = 0; kc < 2; ++kc)
    af[kc] = load_a_frag(sX + (wave * 16) * LDA + kc * 32, LDA);

  size_t obase = (size_t)bh * S * 64;
#pragma unroll
  for (int w = 0; w < 3; ++w) {
    v8f acc[4];
#pragma unroll
    for (int nt = 0; nt < 4; ++nt) acc[nt] = v8f_zero();
#pragma unroll
    for (int kc = 0; kc < 2; ++kc)
#pragma unroll
      for (int nt = 0; nt < 4; ++nt) {
        v16h bf = load_bT_frag(sW + (w * 64 + nt * 16) * LDA + kc * 32, LDA);
        acc[nt] = wmma16(af[kc], bf, acc[nt]);
      }
    const float* bias = (w == 0 ? bq : (w == 1 ? bk : bv)) + h * 64;
    _Float16* outp = (w == 0 ? q : (w == 1 ? k : v));
#pragma unroll
    for (int nt = 0; nt < 4; ++nt)
#pragma unroll
      for (int r = 0; r < 8; ++r) {
        int m = r + 8 * half;
        int c = nt * 16 + ln;
        float val = acc[nt][r] + bias[c];
        if (w == 0) val *= 0.125f;   // fold 1/sqrt(64) into q
        outp[obase + (size_t)(s0 + wave * 16 + m) * 64 + c] = (_Float16)val;
      }
  }
}

// ---------------------------------------------------------------- flash attention
// grid: (S/64, B*H), block 128 (4 waves; wave owns 16 query rows, online softmax).
__global__ void __launch_bounds__(128) attn_kernel(
    const _Float16* __restrict__ q, const _Float16* __restrict__ k,
    const _Float16* __restrict__ v,
    const float* __restrict__ x, float* __restrict__ middle,
    int B, int H, int S, int D) {
  const int LDC = 72;
  __shared__ _Float16 sk[64 * LDC];
  __shared__ _Float16 sv[64 * LDC];
  __shared__ _Float16 sp[4 * 16 * 64];
  int bh = blockIdx.y;
  int b  = bh / H, h = bh % H;
  int s0 = blockIdx.x * 64;
  int tid = threadIdx.x, wave = tid >> 5, lane = tid & 31, half = lane >> 4, ln = lane & 15;
  size_t base = (size_t)bh * S * 64;
  int qrow = s0 + wave * 16;

  v16h qf[2];
#pragma unroll
  for (int kc = 0; kc < 2; ++kc)
    qf[kc] = load_a_frag(q + base + (size_t)qrow * 64 + kc * 32, 64);

  v8f o[4];
  float rmax[8], rsum[8];
#pragma unroll
  for (int t = 0; t < 4; ++t) o[t] = v8f_zero();
#pragma unroll
  for (int r = 0; r < 8; ++r) { rmax[r] = -1e30f; rsum[r] = 0.f; }

  _Float16* myp = sp + wave * 16 * 64;

  for (int t0 = 0; t0 < S; t0 += 64) {
    if (wave == 0) {                             // TDM-stage K and V chunks
      tdm_load_tile(k + base + (size_t)t0 * 64, sk, 64, 64);
      tdm_load_tile(v + base + (size_t)t0 * 64, sv, 64, 64);
      tdm_wait();
    }
    __syncthreads();

    // scores: 16 x 64 for this wave (K = 64 channels)
    v8f s[4];
#pragma unroll
    for (int nt = 0; nt < 4; ++nt) s[nt] = v8f_zero();
#pragma unroll
    for (int kc = 0; kc < 2; ++kc)
#pragma unroll
      for (int nt = 0; nt < 4; ++nt) {
        v16h bf = load_bT_frag(sk + (nt * 16) * LDC + kc * 32, LDC);
        s[nt] = wmma16(qf[kc], bf, s[nt]);
      }

    // online softmax per row (row m = r + 8*half lives in one 16-lane half)
#pragma unroll
    for (int r = 0; r < 8; ++r) {
      float mx = fmaxf(fmaxf(s[0][r], s[1][r]), fmaxf(s[2][r], s[3][r]));
#pragma unroll
      for (int off = 1; off < 16; off <<= 1) mx = fmaxf(mx, __shfl_xor(mx, off, 32));
      float nm   = fmaxf(rmax[r], mx);
      float corr = __expf(rmax[r] - nm);
      rmax[r] = nm;
      float ls = 0.f;
#pragma unroll
      for (int nt = 0; nt < 4; ++nt) {
        float pv = __expf(s[nt][r] - nm);
        ls += pv;
        myp[(r + 8 * half) * 64 + nt * 16 + ln] = (_Float16)pv;
      }
#pragma unroll
      for (int off = 1; off < 16; off <<= 1) ls += __shfl_xor(ls, off, 32);
      rsum[r] = rsum[r] * corr + ls;
#pragma unroll
      for (int nt = 0; nt < 4; ++nt) o[nt][r] *= corr;
    }

    // wave-local LDS RAW: wait for p stores before re-reading as A-fragments
    asm volatile("s_wait_dscnt 0" ::: "memory");

#pragma unroll
    for (int kc = 0; kc < 2; ++kc) {
      v16h pf = load_a_frag(myp + kc * 32, 64);
#pragma unroll
      for (int ct = 0; ct < 4; ++ct) {
        v16h bf = load_bKN_frag(sv, kc * 32, ct * 16, LDC);
        o[ct] = wmma16(pf, bf, o[ct]);
      }
    }
    __syncthreads();
  }

  // middle = x + attn (write fused)
#pragma unroll
  for (int ct = 0; ct < 4; ++ct)
#pragma unroll
    for (int r = 0; r < 8; ++r) {
      int m = r + 8 * half;
      int srow = s0 + wave * 16 + m;
      int c = ct * 16 + ln;
      size_t gi = ((size_t)b * S + srow) * D + h * 64 + c;
      middle[gi] = x[gi] + o[ct][r] / rsum[r];
    }
}

// ---------------------------------------------------------------- tiled GEMM
// C[M][N] = act(A[M][K] * BT[N][K]^T + bias) (+ residual). BM=BN=128, BK=64.
// block 256 (8 waves; 4x2 wave grid; wave = 32x64 of C = 2x4 WMMA tiles).
template <int ACT, int OUTF32>
__global__ void __launch_bounds__(256) gemm_kernel(
    const _Float16* __restrict__ A, const _Float16* __restrict__ BT,
    const float* __restrict__ bias, const float* __restrict__ resid,
    void* __restrict__ outv, int M, int N, int K) {
  const int BK = 64, LDT = BK + 8;
  __shared__ _Float16 sA[128 * LDT];
  __shared__ _Float16 sB[128 * LDT];
  int m0 = blockIdx.y * 128, n0 = blockIdx.x * 128;
  int tid = threadIdx.x, wave = tid >> 5, lane = tid & 31, half = lane >> 4, ln = lane & 15;
  int wrow = (wave >> 1) * 32;
  int wcol = (wave & 1) * 64;

  v8f acc[2][4];
#pragma unroll
  for (int mt = 0; mt < 2; ++mt)
#pragma unroll
    for (int nt = 0; nt < 4; ++nt) acc[mt][nt] = v8f_zero();

  int kSteps = K / BK;
  for (int ks = 0; ks < kSteps; ++ks) {
    int k0 = ks * BK;
    if (wave == 0) {                             // TDM-stage A and B tiles
      tdm_load_tile(A  + (size_t)m0 * K + k0, sA, 128, K);
      tdm_load_tile(BT + (size_t)n0 * K + k0, sB, 128, K);
      tdm_wait();
    }
    if (ks + 1 < kSteps) {                       // hint next K-tile -> global_prefetch_b8
      __builtin_prefetch(A  + (size_t)(m0 + (tid >> 1)) * K + k0 + BK, 0, 0);
      __builtin_prefetch(BT + (size_t)(n0 + (tid >> 1)) * K + k0 + BK, 0, 0);
    }
    __syncthreads();
#pragma unroll
    for (int kc = 0; kc < 2; ++kc) {
      v16h af[2];
#pragma unroll
      for (int mt = 0; mt < 2; ++mt)
        af[mt] = load_a_frag(sA + (wrow + mt * 16) * LDT + kc * 32, LDT);
#pragma unroll
      for (int nt = 0; nt < 4; ++nt) {
        v16h bf = load_bT_frag(sB + (wcol + nt * 16) * LDT + kc * 32, LDT);
        acc[0][nt] = wmma16(af[0], bf, acc[0][nt]);
        acc[1][nt] = wmma16(af[1], bf, acc[1][nt]);
      }
    }
    __syncthreads();
  }

#pragma unroll
  for (int mt = 0; mt < 2; ++mt)
#pragma unroll
    for (int nt = 0; nt < 4; ++nt)
#pragma unroll
      for (int r = 0; r < 8; ++r) {
        int gm = m0 + wrow + mt * 16 + r + 8 * half;
        int gn = n0 + wcol + nt * 16 + ln;
        float val = acc[mt][nt][r] + bias[gn];
        if (ACT) val = 0.5f * val * (1.f + erff(val * 0.70710678118654752f));
        size_t gi = (size_t)gm * N + gn;
        if (OUTF32) ((float*)outv)[gi] = val + resid[gi];
        else        ((_Float16*)outv)[gi] = (_Float16)val;
      }
}

// ---------------------------------------------------------------- host orchestration
extern "C" void kernel_launch(void* const* d_in, const int* in_sizes, int n_in,
                              void* d_out, int out_size, void* d_ws, size_t ws_size,
                              hipStream_t stream) {
  (void)in_sizes; (void)n_in; (void)out_size; (void)ws_size;
  const int B = 8, S = 1024, D = 768, H = 12, C = 64, F = 3072;
  const int M = B * S;

  const float* x    = (const float*)d_in[0];
  const float* ln_g = (const float*)d_in[1];
  const float* ln_b = (const float*)d_in[2];
  const float* Wq   = (const float*)d_in[3];
  const float* bq   = (const float*)d_in[4];
  const float* Wk   = (const float*)d_in[5];
  const float* bk   = (const float*)d_in[6];
  const float* Wv   = (const float*)d_in[7];
  const float* bv   = (const float*)d_in[8];
  const float* W1   = (const float*)d_in[9];
  const float* b1   = (const float*)d_in[10];
  const float* Wm   = (const float*)d_in[11];
  const float* bm   = (const float*)d_in[12];
  const float* W2   = (const float*)d_in[13];
  const float* b2   = (const float*)d_in[14];

  char* ws = (char*)d_ws;
  size_t off = 0;
  auto alloc = [&](size_t bytes) -> void* {
    void* p = ws + off;
    off = (off + bytes + 255) & ~(size_t)255;
    return p;
  };
  _Float16* n1h = (_Float16*)alloc((size_t)M * D * 2);
  _Float16* n2h = (_Float16*)alloc((size_t)M * D * 2);
  _Float16* qh  = (_Float16*)alloc((size_t)M * D * 2);
  _Float16* kh  = (_Float16*)alloc((size_t)M * D * 2);
  _Float16* vh  = (_Float16*)alloc((size_t)M * D * 2);
  float*    mid = (float*)   alloc((size_t)M * D * 4);
  _Float16* h1  = (_Float16*)alloc((size_t)M * F * 2);
  _Float16* h2  = (_Float16*)alloc((size_t)M * F * 2);
  _Float16* W1T = (_Float16*)alloc((size_t)D * F * 2);
  _Float16* WmT = (_Float16*)alloc((size_t)F * F * 2);
  _Float16* W2T = (_Float16*)alloc((size_t)F * D * 2);
  _Float16* WqT = (_Float16*)alloc((size_t)H * C * C * 2);
  _Float16* WkT = (_Float16*)alloc((size_t)H * C * C * 2);
  _Float16* WvT = (_Float16*)alloc((size_t)H * C * C * 2);

  convert_T_kernel<<<512, 256, 0, stream>>>(W1, W1T, 1, D, F);
  convert_T_kernel<<<1024, 256, 0, stream>>>(Wm, WmT, 1, F, F);
  convert_T_kernel<<<512, 256, 0, stream>>>(W2, W2T, 1, F, D);
  convert_T_kernel<<<48, 256, 0, stream>>>(Wq, WqT, H, C, C);
  convert_T_kernel<<<48, 256, 0, stream>>>(Wk, WkT, H, C, C);
  convert_T_kernel<<<48, 256, 0, stream>>>(Wv, WvT, H, C, C);

  ln_kernel<<<M, 256, 0, stream>>>(x, ln_g, ln_b, n1h, D);

  qkv_kernel<<<dim3(S / 64, B * H), 128, 0, stream>>>(n1h, WqT, WkT, WvT, bq, bk, bv,
                                                      qh, kh, vh, B, H, S, D);

  attn_kernel<<<dim3(S / 64, B * H), 128, 0, stream>>>(qh, kh, vh, x, mid, B, H, S, D);

  ln_kernel<<<M, 256, 0, stream>>>(mid, ln_g, ln_b, n2h, D);

  gemm_kernel<1, 0><<<dim3(F / 128, M / 128), 256, 0, stream>>>(n2h, W1T, b1, nullptr, h1, M, F, D);
  gemm_kernel<1, 0><<<dim3(F / 128, M / 128), 256, 0, stream>>>(h1, WmT, bm, nullptr, h2, M, F, F);
  gemm_kernel<0, 1><<<dim3(D / 128, M / 128), 256, 0, stream>>>(h2, W2T, b2, mid, d_out, M, D, F);
}
